// GAT_12232066859281
// MI455X (gfx1250) — compile-verified, matrix operand order checked by reading, not run on hardware
//
#include <hip/hip_runtime.h>
#include <hip/hip_bf16.h>

// ---------------------------------------------------------------------------
// GATv2 (2 layers) for MI455X / gfx1250.
//  - Dense node transforms: V_WMMA_F32_16X16X32_BF16, fully unrolled K,
//    branch-free b128 loads, pre-packed B fragments.
//  - Edge phases: bandwidth/atomic bound scatter-gather living in 192MB L2.
// ---------------------------------------------------------------------------

#define NN   100000
#define EE   1600000
#define INF_ 128
#define EDIM 16
#define HIDC 32
#define OUTC 32
#define HC1  64   // HEADS*HID for layer 1

typedef __attribute__((ext_vector_type(16))) __bf16 v16bf;
typedef __attribute__((ext_vector_type(8)))  float  v8f;

static __device__ __forceinline__ unsigned short f2bf_u16(float f) {
    unsigned u = __float_as_uint(f);
    unsigned r = (u + 0x7FFFu + ((u >> 16) & 1u)) >> 16;   // RNE
    return (unsigned short)r;
}
static __device__ __forceinline__ __bf16 f2bf(float f) {
    return __builtin_bit_cast(__bf16, f2bf_u16(f));
}

// float -> order-preserving uint (for atomic segment max)
static __device__ __forceinline__ unsigned f2ord(float f) {
    unsigned u = __float_as_uint(f);
    return (u & 0x80000000u) ? ~u : (u | 0x80000000u);
}
static __device__ __forceinline__ float ord2f(unsigned t) {
    unsigned u = (t & 0x80000000u) ? (t & 0x7FFFFFFFu) : ~t;
    return __uint_as_float(u);
}
#define ORD_NEG_INF 0x007FFFFFu   // f2ord(-inf)

// ---------------------------------------------------------------------------
// small helpers
// ---------------------------------------------------------------------------
__global__ void init_f32_kernel(float* __restrict__ p, int n, float v) {
    int i = blockIdx.x * blockDim.x + threadIdx.x;
    if (i < n) p[i] = v;
}

__global__ void init_u32_kernel(unsigned* __restrict__ p, int n, unsigned v) {
    int i = blockIdx.x * blockDim.x + threadIdx.x;
    if (i < n) p[i] = v;
}

// elementwise: p = elu(p + bias[col])
__global__ void bias_elu_kernel(float* __restrict__ p, const float* __restrict__ bias,
                                int total, int HC) {
    int i = blockIdx.x * blockDim.x + threadIdx.x;
    if (i >= total) return;
    float v = p[i] + bias[i % HC];
    p[i] = v > 0.0f ? v : expm1f(v);
}

// ---------------------------------------------------------------------------
// Pack a f32 weight matrix W[K,Ncols] into per-lane WMMA B-fragment order:
//   P[ct][ks][lane][e]  (bf16), with
//   kk  = ks*32 + (lane/16)*8 + (e/8)*16 + (e%8)
//   col = ct*16 + lane%16
// so each lane's 16-half fragment is one contiguous 32-byte chunk.
// ---------------------------------------------------------------------------
__global__ void pack_w_kernel(const float* __restrict__ W,
                              unsigned short* __restrict__ P,
                              int K, int Ncols) {
    int idx = blockIdx.x * blockDim.x + threadIdx.x;
    if (idx >= K * Ncols) return;
    int e    = idx & 15;
    int lane = (idx >> 4) & 31;
    int rest = idx >> 9;
    int KS   = K >> 5;
    int ks   = rest % KS;
    int ct   = rest / KS;
    int half = lane >> 4, l16 = lane & 15;
    int kk   = ks * 32 + half * 8 + ((e >> 3) << 4) + (e & 7);
    int col  = ct * 16 + l16;
    P[idx] = f2bf_u16(W[(size_t)kk * Ncols + col]);
}

// ---------------------------------------------------------------------------
// WMMA GEMM: C[M, NT*16] = A[M,K](f32->bf16) @ W(packed bf16 fragments) + bias
// One wave computes a full 16 x (NT*16) row-stripe; 8 waves per block.
// K compile-time (multiple of 32) -> fully unrolled, branch-free loads.
// ---------------------------------------------------------------------------
template <int KDIM, int NT>
__global__ void gemm_bf16_wmma_kernel(const float* __restrict__ A,
                                      const unsigned short* __restrict__ Wp,
                                      const float* __restrict__ bias,
                                      float* __restrict__ Cout, int M) {
    constexpr int KSTEPS = KDIM / 32;
    constexpr int NCOLS  = NT * 16;
    const int lane = threadIdx.x & 31;
    const int wave = threadIdx.x >> 5;
    const int half = lane >> 4;
    const int l16  = lane & 15;
    const int rowTile = (blockIdx.x * (blockDim.x >> 5) + wave) * 16;
    const int arow = rowTile + l16;
    const int ar   = arow < M ? arow : (M - 1);   // clamp: loads always in-bounds

    const v16bf* __restrict__ Wf = (const v16bf*)Wp;  // [NT][KSTEPS][32]

    v8f acc[NT];
#pragma unroll
    for (int t = 0; t < NT; ++t) { v8f z = {}; acc[t] = z; }

#pragma unroll
    for (int ks = 0; ks < KSTEPS; ++ks) {
        const float* ab = A + (size_t)ar * KDIM + ks * 32 + half * 8;
        float4 a0 = ((const float4*)ab)[0];
        float4 a1 = ((const float4*)ab)[1];
        float4 a2 = ((const float4*)(ab + 16))[0];
        float4 a3 = ((const float4*)(ab + 16))[1];
        v16bf af;
        af[0]  = f2bf(a0.x); af[1]  = f2bf(a0.y); af[2]  = f2bf(a0.z); af[3]  = f2bf(a0.w);
        af[4]  = f2bf(a1.x); af[5]  = f2bf(a1.y); af[6]  = f2bf(a1.z); af[7]  = f2bf(a1.w);
        af[8]  = f2bf(a2.x); af[9]  = f2bf(a2.y); af[10] = f2bf(a2.z); af[11] = f2bf(a2.w);
        af[12] = f2bf(a3.x); af[13] = f2bf(a3.y); af[14] = f2bf(a3.z); af[15] = f2bf(a3.w);
#pragma unroll
        for (int t = 0; t < NT; ++t) {
            v16bf wf = Wf[(t * KSTEPS + ks) * 32 + lane];
            acc[t] = __builtin_amdgcn_wmma_f32_16x16x32_bf16(
                /*neg_a=*/false, af, /*neg_b=*/false, wf,
                /*c_mod=*/(short)0, acc[t], /*reuse_a=*/false, /*reuse_b=*/false);
        }
    }

#pragma unroll
    for (int t = 0; t < NT; ++t) {
        int col  = t * 16 + l16;
        float bv = bias[col];
#pragma unroll
        for (int i = 0; i < 8; ++i) {
            int r = rowTile + half * 8 + i;
            if (r < M) Cout[(size_t)r * NCOLS + col] = acc[t][i] + bv;
        }
    }
}

// ---------------------------------------------------------------------------
// Phase 1: per-edge attention logits e[E,H] + segment max via atomicMax.
// 16 lanes cooperate per edge; edge-attr GEMM (K=16) fused from LDS weights.
// ---------------------------------------------------------------------------
template <int H, int C>
__global__ void edge_score_kernel(const float* __restrict__ xl,
                                  const float* __restrict__ xr,
                                  const float* __restrict__ eattr,
                                  const int* __restrict__ ei,
                                  const float* __restrict__ We,   // [16, H*C] f32
                                  const float* __restrict__ att,  // [H*C] f32
                                  float* __restrict__ eout,       // [E,H]
                                  unsigned* __restrict__ emax,    // [N,H] ordered-uint
                                  int E) {
    constexpr int HC   = H * C;
    constexpr int VALS = HC / 16;
    __shared__ float Wsh[EDIM * HC];
    __shared__ float Ash[HC];
    for (int i = threadIdx.x; i < EDIM * HC; i += blockDim.x) Wsh[i] = We[i];
    for (int i = threadIdx.x; i < HC; i += blockDim.x) Ash[i] = att[i];
    __syncthreads();

    int gid  = blockIdx.x * blockDim.x + threadIdx.x;
    int edge = gid >> 4;
    int l16  = gid & 15;
    if (edge >= E) return;
    int s = ei[edge];
    int d = ei[E + edge];

    // load the 16 edge attributes (one 64B row, L2-friendly)
    float ek[16];
    const float4* ap = (const float4*)(eattr + (size_t)edge * EDIM);
#pragma unroll
    for (int q = 0; q < 4; ++q) {
        float4 v = ap[q];
        ek[4 * q + 0] = v.x; ek[4 * q + 1] = v.y;
        ek[4 * q + 2] = v.z; ek[4 * q + 3] = v.w;
    }

    float hs[H];
#pragma unroll
    for (int h = 0; h < H; ++h) hs[h] = 0.0f;

#pragma unroll
    for (int v = 0; v < VALS; ++v) {
        int c = l16 + 16 * v;
        float ea = 0.0f;
#pragma unroll
        for (int k = 0; k < EDIM; ++k) ea += ek[k] * Wsh[k * HC + c];
        float m  = xl[(size_t)s * HC + c] + xr[(size_t)d * HC + c] + ea;
        float lr = m > 0.0f ? m : 0.2f * m;
        hs[(16 * v) / C] += lr * Ash[c];   // head index is unroll-constant
    }
#pragma unroll
    for (int h = 0; h < H; ++h)
        for (int off = 8; off > 0; off >>= 1)
            hs[h] += __shfl_xor(hs[h], off, 16);

    if (l16 == 0) {
#pragma unroll
        for (int h = 0; h < H; ++h) {
            eout[(size_t)edge * H + h] = hs[h];
            atomicMax(&emax[(size_t)d * H + h], f2ord(hs[h]));
        }
    }
}

// ---------------------------------------------------------------------------
// Phase 2: ee = exp(e - emax[dst]); denom[dst] += ee  (segment softmax denom)
// ---------------------------------------------------------------------------
template <int H>
__global__ void edge_softmax_kernel(float* __restrict__ eout,
                                    const int* __restrict__ ei,
                                    const unsigned* __restrict__ emax,
                                    float* __restrict__ denom, int E) {
    int idx = blockIdx.x * blockDim.x + threadIdx.x;
    if (idx >= E * H) return;
    int edge = idx / H;
    int h    = idx % H;
    int d    = ei[E + edge];
    float em = ord2f(emax[(size_t)d * H + h]);
    if (!__builtin_isfinite(em)) em = 0.0f;
    float ee = expf(eout[idx] - em);
    eout[idx] = ee;
    atomicAdd(&denom[(size_t)d * H + h], ee);
}

// ---------------------------------------------------------------------------
// Phase 3: out[dst] += xl[src] * alpha, alpha = ee / (denom[dst] + 1e-16)
// ---------------------------------------------------------------------------
template <int H, int C>
__global__ void edge_aggregate_kernel(const float* __restrict__ xl,
                                      const float* __restrict__ eout,
                                      const float* __restrict__ denom,
                                      const int* __restrict__ ei,
                                      float* __restrict__ out, int E) {
    constexpr int HC   = H * C;
    constexpr int VALS = HC / 16;
    int gid  = blockIdx.x * blockDim.x + threadIdx.x;
    int edge = gid >> 4;
    int l16  = gid & 15;
    if (edge >= E) return;
    int s = ei[edge];
    int d = ei[E + edge];
#pragma unroll
    for (int v = 0; v < VALS; ++v) {
        int c = l16 + 16 * v;
        int h = (16 * v) / C;
        float ee    = eout[(size_t)edge * H + h];
        float den   = denom[(size_t)d * H + h];
        float alpha = ee / (den + 1e-16f);
        atomicAdd(&out[(size_t)d * HC + c], xl[(size_t)s * HC + c] * alpha);
    }
}

// ---------------------------------------------------------------------------
// launcher
// ---------------------------------------------------------------------------
static inline int cdiv(long a, long b) { return (int)((a + b - 1) / b); }

extern "C" void kernel_launch(void* const* d_in, const int* in_sizes, int n_in,
                              void* d_out, int out_size, void* d_ws, size_t ws_size,
                              hipStream_t stream) {
    (void)in_sizes; (void)n_in; (void)out_size; (void)ws_size;

    const float* x     = (const float*)d_in[0];
    const int*   ei    = (const int*)d_in[1];
    const float* eatt  = (const float*)d_in[2];
    const float* W1l   = (const float*)d_in[3];
    const float* b1l   = (const float*)d_in[4];
    const float* W1r   = (const float*)d_in[5];
    const float* b1r   = (const float*)d_in[6];
    const float* W1e   = (const float*)d_in[7];
    const float* att1  = (const float*)d_in[8];
    const float* bias1 = (const float*)d_in[9];
    const float* W2l   = (const float*)d_in[10];
    const float* b2l   = (const float*)d_in[11];
    const float* W2r   = (const float*)d_in[12];
    const float* b2r   = (const float*)d_in[13];
    const float* W2e   = (const float*)d_in[14];
    const float* att2  = (const float*)d_in[15];
    const float* bias2 = (const float*)d_in[16];
    const float* Wlin  = (const float*)d_in[17];
    const float* blin  = (const float*)d_in[18];
    float* out = (float*)d_out;

    // ---- workspace carve-up (bytes) ----
    char* ws = (char*)d_ws;
    const size_t SZ_XL1  = (size_t)NN * HC1 * 4;   // 25.6 MB
    const size_t SZ_E1   = (size_t)EE * 2 * 4;     // 12.8 MB
    const size_t SZ_M1   = (size_t)NN * 2 * 4;     // 0.8 MB
    const size_t SZ_ACC2 = (size_t)NN * HIDC * 4;  // 12.8 MB
    size_t off = 0;
    float*    xl1  = (float*)(ws + off); off += SZ_XL1;   // layer2 reuses as xl2
    float*    xr1  = (float*)(ws + off); off += SZ_XL1;   // layer2 reuses as xr2
    float*    acc1 = (float*)(ws + off); off += SZ_XL1;   // becomes h1 in-place
    float*    e1   = (float*)(ws + off); off += SZ_E1;    // layer2 reuses as e2
    unsigned* emax = (unsigned*)(ws + off); off += SZ_M1; // reused by layer2
    float*    den  = (float*)(ws + off); off += SZ_M1;    // reused by layer2
    float*    acc2 = (float*)(ws + off); off += SZ_ACC2;  // becomes h2 in-place
    unsigned short* wbf = (unsigned short*)(ws + off);    // 256B-aligned region
    unsigned short* w1l_p = wbf + 0;       // 128*64
    unsigned short* w1r_p = wbf + 8192;    // 128*64
    unsigned short* w2l_p = wbf + 16384;   // 64*32
    unsigned short* w2r_p = wbf + 18432;   // 64*32
    unsigned short* wln_p = wbf + 20480;   // 32*32

    const int TB = 256;

    // ---- pack dense-GEMM weights into bf16 WMMA fragment order (tiny) ----
    pack_w_kernel<<<cdiv(8192, TB), TB, 0, stream>>>(W1l, w1l_p, INF_, HC1);
    pack_w_kernel<<<cdiv(8192, TB), TB, 0, stream>>>(W1r, w1r_p, INF_, HC1);
    pack_w_kernel<<<cdiv(2048, TB), TB, 0, stream>>>(W2l, w2l_p, HC1, HIDC);
    pack_w_kernel<<<cdiv(2048, TB), TB, 0, stream>>>(W2r, w2r_p, HC1, HIDC);
    pack_w_kernel<<<cdiv(1024, TB), TB, 0, stream>>>(Wlin, wln_p, HIDC, OUTC);

    const int gemmBlocks = cdiv(NN, 16 * 8);   // 8 waves/block, 16 rows/wave

    // ================= layer 1 (H=2, C=32, concat) =================
    gemm_bf16_wmma_kernel<INF_, HC1 / 16>
        <<<gemmBlocks, TB, 0, stream>>>(x, w1l_p, b1l, xl1, NN);
    gemm_bf16_wmma_kernel<INF_, HC1 / 16>
        <<<gemmBlocks, TB, 0, stream>>>(x, w1r_p, b1r, xr1, NN);

    init_u32_kernel<<<cdiv(NN * 2, TB), TB, 0, stream>>>(emax, NN * 2, ORD_NEG_INF);
    init_f32_kernel<<<cdiv(NN * 2, TB), TB, 0, stream>>>(den, NN * 2, 0.0f);
    init_f32_kernel<<<cdiv((long)NN * HC1, TB), TB, 0, stream>>>(acc1, NN * HC1, 0.0f);

    {
        int nthr = EE * 16;
        edge_score_kernel<2, 32><<<cdiv(nthr, TB), TB, 0, stream>>>(
            xl1, xr1, eatt, ei, W1e, att1, e1, emax, EE);
        edge_softmax_kernel<2><<<cdiv((long)EE * 2, TB), TB, 0, stream>>>(
            e1, ei, emax, den, EE);
        edge_aggregate_kernel<2, 32><<<cdiv(nthr, TB), TB, 0, stream>>>(
            xl1, e1, den, ei, acc1, EE);
    }
    bias_elu_kernel<<<cdiv((long)NN * HC1, TB), TB, 0, stream>>>(acc1, bias1,
                                                                NN * HC1, HC1);
    // acc1 is now h1 [N,64]

    // ================= layer 2 (H=1, C=32, mean==identity) =================
    gemm_bf16_wmma_kernel<HC1, HIDC / 16>
        <<<gemmBlocks, TB, 0, stream>>>(acc1, w2l_p, b2l, xl1, NN);
    gemm_bf16_wmma_kernel<HC1, HIDC / 16>
        <<<gemmBlocks, TB, 0, stream>>>(acc1, w2r_p, b2r, xr1, NN);

    init_u32_kernel<<<cdiv(NN, TB), TB, 0, stream>>>(emax, NN, ORD_NEG_INF);
    init_f32_kernel<<<cdiv(NN, TB), TB, 0, stream>>>(den, NN, 0.0f);
    init_f32_kernel<<<cdiv((long)NN * HIDC, TB), TB, 0, stream>>>(acc2, NN * HIDC, 0.0f);

    {
        int nthr = EE * 16;
        edge_score_kernel<1, 32><<<cdiv(nthr, TB), TB, 0, stream>>>(
            xl1, xr1, eatt, ei, W2e, att2, e1, emax, EE);
        edge_softmax_kernel<1><<<cdiv((long)EE, TB), TB, 0, stream>>>(
            e1, ei, emax, den, EE);
        edge_aggregate_kernel<1, 32><<<cdiv(nthr, TB), TB, 0, stream>>>(
            xl1, e1, den, ei, acc2, EE);
    }
    bias_elu_kernel<<<cdiv((long)NN * HIDC, TB), TB, 0, stream>>>(acc2, bias2,
                                                                 NN * HIDC, HIDC);
    // acc2 is now h2 [N,32]

    // ================= final linear: out = h2 @ Wlin + blin =================
    gemm_bf16_wmma_kernel<HIDC, OUTC / 16>
        <<<gemmBlocks, TB, 0, stream>>>(acc2, wln_p, blin, out, NN);
}
